// SelfAttentionChannel_70772471103574
// MI455X (gfx1250) — matrix operand order repaired and checked by reading, not compiled, not run on hardware
//
#include <hip/hip_runtime.h>
#include <hip/hip_bf16.h>

typedef __attribute__((ext_vector_type(16))) __bf16 bf16x16;
typedef __attribute__((ext_vector_type(8)))  float  f32x8;

#define TILE_K    32
#define LDSPITCH  40      // ushorts per LDS row: 32 data + 8 pad (80B rows keep 16B alignment)
#define NTHREADS  256     // 8 wave32s
#define KSPLIT    4       // split-K factor for the QQᵀ GEMM (occupancy)
#define SPART     ((size_t)16 * 256 * 256)   // elements per split-K partial of S

__device__ __forceinline__ unsigned short f32_to_bf16_rne(float f) {
  unsigned u = __float_as_uint(f);
  u += 0x7FFFu + ((u >> 16) & 1u);   // round-to-nearest-even
  return (unsigned short)(u >> 16);
}

union FragCast {
  uint4 u[2];
  bf16x16 v;
};

// A fragment: 16x32 bf16, lane L -> row base+(L&15); K = (L>>4)*8 + (j/8)*16 + j%8
__device__ __forceinline__ bf16x16 load_frag_A(const unsigned short* lds, int row16, int lane) {
  const unsigned short* p = lds + (row16 + (lane & 15)) * LDSPITCH;
  const int k0 = (lane >> 4) << 3;            // 0 or 8
  FragCast f;
  f.u[0] = *reinterpret_cast<const uint4*>(p + k0);        // K = k0 .. k0+7
  f.u[1] = *reinterpret_cast<const uint4*>(p + k0 + 16);   // K = k0+16 .. k0+23
  return f.v;
}

// B fragment: 32x16 bf16 staged as LDS[n][k]; lane L -> col base+(L&15); K = (L>>4)*8 + (j/8)*16 + j%8
__device__ __forceinline__ bf16x16 load_frag_B(const unsigned short* lds, int col16, int lane) {
  const unsigned short* p = lds + (col16 + (lane & 15)) * LDSPITCH + ((lane >> 4) << 3);
  FragCast f;
  f.u[0] = *reinterpret_cast<const uint4*>(p);
  f.u[1] = *reinterpret_cast<const uint4*>(p + 16);
  return f.v;
}

// Async copy of one 16B chunk into LDS (CDNA5 GLOBAL_LOAD_ASYNC_TO_LDS_B128,
// tracked by ASYNCcnt). Per-lane global address + per-lane LDS byte offset.
__device__ __forceinline__ void async_copy_b128(const unsigned short* gsrc,
                                                const unsigned short* ldst) {
  const unsigned long long ga = (unsigned long long)(const void*)gsrc;
  const unsigned lo = (unsigned)(unsigned long long)(const void*)ldst; // low 32b = LDS offset
  asm volatile("global_load_async_to_lds_b128 %0, %1, off"
               :: "v"(lo), "v"(ga) : "memory");
}

__device__ __forceinline__ void wait_async_zero() {
  asm volatile("s_wait_asynccnt 0x0" ::: "memory");
}

// Shared 128x128 GEMM mainloop over K in steps of 32, all-bf16 operands.
// A panel: row-major [row][k] at Apanel (rows = output M rows), leading dim lda (elements).
// B source: if B_TN != 0, Bsrc is row-major [n][k] already offset to the n-tile;
//           else Bsrc is row-major [k][n] and nbase selects the 128-wide n-column slab.
__device__ __forceinline__ void gemm_core(unsigned short* ldsA, unsigned short* ldsB,
                                          const unsigned short* Apanel, int lda,
                                          const unsigned short* Bsrc, int ldb, int B_TN,
                                          int nbase, int K, int tid, f32x8 acc[2][4]) {
  const int lane = tid & 31;
  const int wave = tid >> 5;
  const int wm = (wave & 3) * 32;   // 4 waves tile M (32 rows each)
  const int wn = (wave >> 2) * 64;  // 2 waves tile N (64 cols each)

  for (int k0 = 0; k0 < K; k0 += TILE_K) {
    // ---- stage A: 128x32 bf16 panel (8KB) via async-to-LDS, 16B per lane, 2 per thread
    #pragma unroll
    for (int i = 0; i < 2; ++i) {
      const int chunk = tid + i * NTHREADS;     // 512 x 16B chunks
      const int row = chunk >> 2;
      const int s8  = (chunk & 3) << 3;         // 8-element segment within the 32-wide row
      async_copy_b128(Apanel + row * lda + k0 + s8, ldsA + row * LDSPITCH + s8);
    }
    // ---- stage B -> LDS[n][k]
    if (B_TN) {
      #pragma unroll
      for (int i = 0; i < 2; ++i) {
        const int chunk = tid + i * NTHREADS;
        const int row = chunk >> 2;
        const int s8  = (chunk & 3) << 3;
        async_copy_b128(Bsrc + row * ldb + k0 + s8, ldsB + row * LDSPITCH + s8);
      }
    } else {
      // [k][n] global -> [n][k] LDS: 16B coalesced loads along n, 2B scattered LDS stores
      #pragma unroll
      for (int i = 0; i < 2; ++i) {
        const int chunk = tid + i * NTHREADS;   // 32 k-rows x 16 chunks of 8 n-elements
        const int krow = chunk >> 4;
        const int n8   = (chunk & 15) << 3;
        const uint4 v = *reinterpret_cast<const uint4*>(Bsrc + (k0 + krow) * (long)ldb + nbase + n8);
        const unsigned short* e = reinterpret_cast<const unsigned short*>(&v);
        #pragma unroll
        for (int t = 0; t < 8; ++t) ldsB[(n8 + t) * LDSPITCH + krow] = e[t];
      }
    }
    wait_async_zero();       // this wave's async-to-LDS chunks have landed
    __syncthreads();         // everyone's chunks visible

    // ---- 2x4 grid of 16x16 WMMA tiles per wave
    const bf16x16 a0 = load_frag_A(ldsA, wm,      lane);
    const bf16x16 a1 = load_frag_A(ldsA, wm + 16, lane);
    #pragma unroll
    for (int ni = 0; ni < 4; ++ni) {
      const bf16x16 b = load_frag_B(ldsB, wn + ni * 16, lane);
      acc[0][ni] = __builtin_amdgcn_wmma_f32_16x16x32_bf16(false, a0, false, b,
                                                           (short)0, acc[0][ni], false, false);
      acc[1][ni] = __builtin_amdgcn_wmma_f32_16x16x32_bf16(false, a1, false, b,
                                                           (short)0, acc[1][ni], false, false);
    }
    __syncthreads();
  }
}

// C/D layout: lane L, reg r -> M = (L>>4)*8 + r, N = L&15 (within each 16x16 tile)
__device__ __forceinline__ void store_acc_f32(f32x8 acc[2][4], float* outp, long ldo,
                                              const float* addp, float scale, int tid) {
  const int lane = tid & 31;
  const int wave = tid >> 5;
  const int wm = (wave & 3) * 32;
  const int wn = (wave >> 2) * 64;
  const int rb = (lane >> 4) * 8;
  const int cb = lane & 15;
  #pragma unroll
  for (int mi = 0; mi < 2; ++mi)
    #pragma unroll
    for (int ni = 0; ni < 4; ++ni)
      #pragma unroll
      for (int r = 0; r < 8; ++r) {
        const long row = wm + mi * 16 + rb + r;
        const long col = wn + ni * 16 + cb;
        float v = acc[mi][ni][r] * scale;
        if (addp) v += addp[row * ldo + col];
        outp[row * ldo + col] = v;
      }
}

__device__ __forceinline__ void store_acc_bf16(f32x8 acc[2][4], unsigned short* outp,
                                               long ldo, int tid) {
  const int lane = tid & 31;
  const int wave = tid >> 5;
  const int wm = (wave & 3) * 32;
  const int wn = (wave >> 2) * 64;
  const int rb = (lane >> 4) * 8;
  const int cb = lane & 15;
  #pragma unroll
  for (int mi = 0; mi < 2; ++mi)
    #pragma unroll
    for (int ni = 0; ni < 4; ++ni)
      #pragma unroll
      for (int r = 0; r < 8; ++r) {
        const long row = wm + mi * 16 + rb + r;
        const long col = wn + ni * 16 + cb;
        outp[row * ldo + col] = f32_to_bf16_rne(acc[mi][ni][r]);
      }
}

__device__ __forceinline__ void zero_acc(f32x8 acc[2][4]) {
  #pragma unroll
  for (int mi = 0; mi < 2; ++mi)
    #pragma unroll
    for (int ni = 0; ni < 4; ++ni)
      #pragma unroll
      for (int r = 0; r < 8; ++r) acc[mi][ni][r] = 0.0f;
}

// K0: fp32 -> bf16 bulk conversion (one float4 -> ushort4 per thread)
__global__ __launch_bounds__(NTHREADS) void k_convert(const float* __restrict__ src,
                                                      unsigned short* __restrict__ dst,
                                                      long n4) {
  const long i = (long)blockIdx.x * NTHREADS + threadIdx.x;
  if (i >= n4) return;
  const float4 v = reinterpret_cast<const float4*>(src)[i];
  ushort4 s;
  s.x = f32_to_bf16_rne(v.x); s.y = f32_to_bf16_rne(v.y);
  s.z = f32_to_bf16_rne(v.z); s.w = f32_to_bf16_rne(v.w);
  reinterpret_cast<ushort4*>(dst)[i] = s;
}

// K1: Spartial[ks][b] = (Q Qᵀ over K-slice ks) * c^-0.5
// grid (2, 2*KSPLIT, 16): x = m-tile, y = {n-tile, k-slice}, z = batch. 256 blocks.
__global__ __launch_bounds__(NTHREADS) void k_qqt(const unsigned short* __restrict__ Qb,
                                                  float* __restrict__ Spartial) {
  __shared__ unsigned short ldsA[128 * LDSPITCH];
  __shared__ unsigned short ldsB[128 * LDSPITCH];
  const int tid = threadIdx.x;
  const int b = blockIdx.z;
  const int ks = blockIdx.y >> 1;               // 0..3 K-slice
  const long mbase = (long)blockIdx.x * 128;
  const long nbase = (long)(blockIdx.y & 1) * 128;
  const long koff  = (long)ks * 1024;           // 4096 / KSPLIT
  const unsigned short* Q = Qb + (size_t)b * 256 * 4096;
  f32x8 acc[2][4];
  zero_acc(acc);
  gemm_core(ldsA, ldsB, Q + mbase * 4096 + koff, 4096, Q + nbase * 4096 + koff, 4096,
            /*B_TN=*/1, 0, 1024, tid, acc);
  store_acc_f32(acc, Spartial + (size_t)ks * SPART + (size_t)b * 65536 + mbase * 256 + nbase,
                256, nullptr, 0.0625f /* 256^-0.5 */, tid);
}

// K2: sum the KSPLIT partials (fixed order -> deterministic), then row softmax -> bf16 P.
// One wave per 256-wide row.
__global__ __launch_bounds__(NTHREADS) void k_softmax(const float* __restrict__ Spartial,
                                                      unsigned short* __restrict__ Pb) {
  const int tid = threadIdx.x, lane = tid & 31, wave = tid >> 5;
  const size_t row = (size_t)blockIdx.x * 8 + wave;
  const float* p0 = Spartial + row * 256;
  unsigned short* q = Pb + row * 256;
  float v[8];
  float mx = -3.402823466e38f;
  #pragma unroll
  for (int i = 0; i < 8; ++i) {
    const size_t idx = lane + i * 32;
    float s = p0[idx];
    #pragma unroll
    for (int ks = 1; ks < KSPLIT; ++ks) s += p0[(size_t)ks * SPART + idx];
    v[i] = s;
    mx = fmaxf(mx, s);
  }
  #pragma unroll
  for (int off = 16; off > 0; off >>= 1) mx = fmaxf(mx, __shfl_xor(mx, off, 32));
  float s = 0.0f;
  #pragma unroll
  for (int i = 0; i < 8; ++i) { v[i] = __expf(v[i] - mx); s += v[i]; }
  #pragma unroll
  for (int off = 16; off > 0; off >>= 1) s += __shfl_xor(s, off, 32);
  const float inv = 1.0f / s;
  #pragma unroll
  for (int i = 0; i < 8; ++i) q[lane + i * 32] = f32_to_bf16_rne(v[i] * inv);
}

// K3: M[b] = W · P[b]  (256x256x256, folds the 1x1 conv ahead of the big GEMM); bf16 out
__global__ __launch_bounds__(NTHREADS) void k_wp(const unsigned short* __restrict__ Wb,
                                                 const unsigned short* __restrict__ Pb,
                                                 unsigned short* __restrict__ Mb) {
  __shared__ unsigned short ldsA[128 * LDSPITCH];
  __shared__ unsigned short ldsB[128 * LDSPITCH];
  const int tid = threadIdx.x;
  const int b = blockIdx.z;
  const long mbase = (long)blockIdx.x * 128;
  const long nbase = (long)blockIdx.y * 128;
  f32x8 acc[2][4];
  zero_acc(acc);
  gemm_core(ldsA, ldsB, Wb + mbase * 256, 256,
            Pb + (size_t)b * 65536, 256, /*B_TN=*/0, (int)nbase, 256, tid, acc);
  store_acc_bf16(acc, Mb + (size_t)b * 65536 + mbase * 256 + nbase, 256, tid);
}

// K4: out[b] = M[b] · V[b] + x_middle[b]  (V = Vb[b] bf16 [256,4096]; residual from f32 xm)
__global__ __launch_bounds__(NTHREADS) void k_mv(const unsigned short* __restrict__ Mb,
                                                 const unsigned short* __restrict__ Vb,
                                                 const float* __restrict__ xm,
                                                 float* __restrict__ out) {
  __shared__ unsigned short ldsA[128 * LDSPITCH];
  __shared__ unsigned short ldsB[128 * LDSPITCH];
  const int tid = threadIdx.x;
  const int b = blockIdx.z;
  const long mbase = (long)blockIdx.x * 128;
  const long nbase = (long)blockIdx.y * 128;
  f32x8 acc[2][4];
  zero_acc(acc);
  gemm_core(ldsA, ldsB, Mb + (size_t)b * 65536 + mbase * 256, 256,
            Vb + (size_t)b * 256 * 4096, 4096, /*B_TN=*/0, (int)nbase, 256, tid, acc);
  store_acc_f32(acc, out + (size_t)b * 1048576 + mbase * 4096 + nbase, 4096,
                xm + (size_t)b * 1048576 + mbase * 4096 + nbase, 1.0f, tid);
}

extern "C" void kernel_launch(void* const* d_in, const int* in_sizes, int n_in,
                              void* d_out, int out_size, void* d_ws, size_t ws_size,
                              hipStream_t stream) {
  const float* x      = (const float*)d_in[0];  // [16,256,64,64]
  const float* xm     = (const float*)d_in[1];  // [16,256,64,64]
  const float* w_conv = (const float*)d_in[2];  // [256,256]
  float* out = (float*)d_out;

  // Workspace layout: Qb 32Mi | Vb 32Mi | Wb 128Ki | Spartial 16Mi | Pb 2Mi | Mb 2Mi
  unsigned short* Qb = (unsigned short*)d_ws;                 // bf16 [16,256,4096]
  unsigned short* Vb = Qb + (size_t)16 * 256 * 4096;          // bf16 [16,256,4096]
  unsigned short* Wb = Vb + (size_t)16 * 256 * 4096;          // bf16 [256,256]
  float*          S  = (float*)(Wb + (size_t)256 * 256);      // f32  [KSPLIT][16,256,256]
  unsigned short* Pb = (unsigned short*)(S + (size_t)KSPLIT * SPART); // bf16 [16,256,256]
  unsigned short* Mb = Pb + (size_t)16 * 256 * 256;           // bf16 [16,256,256]

  const dim3 blk(NTHREADS);
  const long nx4 = (long)16 * 256 * 4096 / 4;   // float4 chunks in x / xm
  const long nw4 = (long)256 * 256 / 4;
  k_convert<<<dim3((unsigned)((nx4 + NTHREADS - 1) / NTHREADS)), blk, 0, stream>>>(x,  Qb, nx4);
  k_convert<<<dim3((unsigned)((nx4 + NTHREADS - 1) / NTHREADS)), blk, 0, stream>>>(xm, Vb, nx4);
  k_convert<<<dim3((unsigned)((nw4 + NTHREADS - 1) / NTHREADS)), blk, 0, stream>>>(w_conv, Wb, nw4);

  k_qqt    <<<dim3(2, 2 * KSPLIT, 16), blk, 0, stream>>>(Qb, S);
  k_softmax<<<dim3(512),               blk, 0, stream>>>(S, Pb);
  k_wp     <<<dim3(2, 2, 16),          blk, 0, stream>>>(Wb, Pb, Mb);
  k_mv     <<<dim3(2, 32, 16),         blk, 0, stream>>>(Mb, Vb, xm, out);
}
